// EnrichedHierarchicalGraphModule_66563403153824
// MI455X (gfx1250) — matrix operand order repaired and checked by reading, not compiled, not run on hardware
//
#include <hip/hip_runtime.h>

static const int NN = 60000;        // total nodes
static const int NU = 30000;        // users / items
static const int EE = 600000;       // edges
static const int NC = 64;           // clusters

typedef __attribute__((ext_vector_type(16))) _Float16 v16h;
typedef __attribute__((ext_vector_type(8)))  float    v8f;
typedef __attribute__((ext_vector_type(4)))  float    v4f;

__device__ __forceinline__ float leaky02(float x){ return x > 0.f ? x : 0.2f * x; }

__device__ __forceinline__ void atomicMaxF(float* addr, float val){
  unsigned int* u = (unsigned int*)addr;
  unsigned int old = *u;
  while (__uint_as_float(old) < val){
    unsigned int assumed = old;
    old = atomicCAS(u, assumed, __float_as_uint(val));
    if (old == assumed) break;
  }
}

// ---------------- utility kernels ----------------
__global__ void fill_k(float* p, float v, long long n){
  long long i = (long long)blockIdx.x * blockDim.x + threadIdx.x;
  if (i < n) p[i] = v;
}
__global__ void copy_k(float* dst, const float* src, long long n){
  long long i = (long long)blockIdx.x * blockDim.x + threadIdx.x;
  if (i < n) dst[i] = src[i];
}
__global__ void rsqrt_k(float* p, int n){
  int i = blockIdx.x * blockDim.x + threadIdx.x;
  if (i < n) p[i] = rsqrtf(p[i]);
}
// Bh[n*K + k] = (f16) B[k*Nc + n]   (transpose + f32->f16, weights only: tiny)
__global__ void convT_k(const float* B, _Float16* Bh, int K, int Nc){
  int t = blockIdx.x * blockDim.x + threadIdx.x;
  if (t >= K * Nc) return;
  int n = t / K, k = t - n * K;
  Bh[t] = (_Float16)B[(long long)k * Nc + n];
}

// ---------------- WMMA GEMM: C[M,Nc] = A[M,K] @ B[K,Nc] (+bias)(+relu) ----------------
// B pre-converted to f16 and transposed: Bh[n*K + k]. One wave per 16x64 C strip
// (4 N-tiles share one A fragment). A loaded as 4x b128 f32 vectors per k-step.
// MODE 0: A row-major stride lda. MODE 1: A row = concat(A[64 cols], A2[32 cols]).
template<int MODE>
__global__ __launch_bounds__(32)
void wmma_gemm(const float* __restrict__ A, int lda, const float* __restrict__ A2,
               const _Float16* __restrict__ Bh, const float* __restrict__ bias,
               float* __restrict__ C, int ldc, int K, int relu){
  int lane = threadIdx.x;
  int hf = lane >> 4;        // half-wave
  int lm = lane & 15;
  int mrow = blockIdx.x * 16 + lm;   // A row held by this lane
  int n0 = blockIdx.y * 64;          // first column of this wave's strip
  v8f acc0 = {}, acc1 = {}, acc2 = {}, acc3 = {};
  for (int kt = 0; kt < K; kt += 32){
    // A 16x32 f16 fragment: elem i -> k = kt+16*hf+i (contiguous, matches B)
    int k0 = kt + 16 * hf;
    int k1 = kt + 16 * hf + 8;
    v4f a0, a1, a2, a3;
    if (MODE == 0){
      const float* p = A + (long long)mrow * lda;
      a0 = *(const v4f*)(p + k0); a1 = *(const v4f*)(p + k0 + 4);
      a2 = *(const v4f*)(p + k1); a3 = *(const v4f*)(p + k1 + 4);
    } else {
      const float* p0 = (k0 < 64) ? A + (long long)mrow * 64 + k0
                                  : A2 + (long long)mrow * 32 + (k0 - 64);
      const float* p1 = (k1 < 64) ? A + (long long)mrow * 64 + k1
                                  : A2 + (long long)mrow * 32 + (k1 - 64);
      a0 = *(const v4f*)p0; a1 = *(const v4f*)(p0 + 4);
      a2 = *(const v4f*)p1; a3 = *(const v4f*)(p1 + 4);
    }
    v16h a;
#pragma unroll
    for (int i = 0; i < 4; ++i){
      a[i]      = (_Float16)a0[i];
      a[i + 4]  = (_Float16)a1[i];
      a[i + 8]  = (_Float16)a2[i];
      a[i + 12] = (_Float16)a3[i];
    }
    // B 32x16 f16 fragment: elem i -> k = kt+16*hf+i, col = lane&15 -> contiguous in Bh
    int kb = kt + 16 * hf;
    const _Float16* bp = Bh + kb;
    v16h b0 = *(const v16h*)(bp + (long long)(n0 + lm) * K);
    v16h b1 = *(const v16h*)(bp + (long long)(n0 + 16 + lm) * K);
    v16h b2 = *(const v16h*)(bp + (long long)(n0 + 32 + lm) * K);
    v16h b3 = *(const v16h*)(bp + (long long)(n0 + 48 + lm) * K);
    acc0 = __builtin_amdgcn_wmma_f32_16x16x32_f16(false, a, false, b0, (short)0, acc0, false, false);
    acc1 = __builtin_amdgcn_wmma_f32_16x16x32_f16(false, a, false, b1, (short)0, acc1, false, false);
    acc2 = __builtin_amdgcn_wmma_f32_16x16x32_f16(false, a, false, b2, (short)0, acc2, false, false);
    acc3 = __builtin_amdgcn_wmma_f32_16x16x32_f16(false, a, false, b3, (short)0, acc3, false, false);
  }
#pragma unroll
  for (int t = 0; t < 4; ++t){
    v8f acc = (t == 0) ? acc0 : (t == 1) ? acc1 : (t == 2) ? acc2 : acc3;
    int n = n0 + t * 16 + lm;
    float bv = bias ? bias[n] : 0.f;   // hoisted: one load per tile, no per-row branch
#pragma unroll
    for (int r = 0; r < 8; ++r){
      int m = blockIdx.x * 16 + r + 8 * hf;  // C layout: VGPR r, lanes 16-31 hold M = r+8
      float v = acc[r] + bv;
      if (relu) v = fmaxf(v, 0.f);
      C[(long long)m * ldc + n] = v;
    }
  }
}

// ---------------- GCN ----------------
__global__ void gcn_deg_k(const int* src, const int* dst, const int* assign, float* deg, int E){
  int e = blockIdx.x * blockDim.x + threadIdx.x;
  if (e >= E) return;
  if (assign[src[e]] == assign[dst[e]]) atomicAdd(&deg[dst[e]], 1.f);
}
__global__ void gcn_agg_k(const int* src, const int* dst, const int* assign,
                          const float* dinv, const float* h, float* agg, int E){
  int e = blockIdx.x * 4 + (threadIdx.x >> 6);
  int f = threadIdx.x & 63;
  if (e >= E) return;
  int s = src[e], d = dst[e];
  if (assign[s] != assign[d]) return;
  float coef = dinv[s] * dinv[d];
  atomicAdd(&agg[(long long)d * 64 + f], coef * h[(long long)s * 64 + f]);
}
__global__ void gcn_final_k(float* Xc, const float* dinv, const float* h, const float* b, int n){
  long long t = (long long)blockIdx.x * blockDim.x + threadIdx.x;
  if (t >= (long long)n * 64) return;
  int node = (int)(t >> 6), d = (int)(t & 63);
  Xc[t] += dinv[node] * dinv[node] * h[t] + b[d];
}

// ---------------- cluster pooling ----------------
__global__ void pool_k(const float* Xc, const int* assign, float* ce, float* cnt, int n){
  long long t = (long long)blockIdx.x * blockDim.x + threadIdx.x;
  if (t >= (long long)n * 64) return;
  int node = (int)(t >> 6), d = (int)(t & 63);
  int c = assign[node];
  atomicAdd(&ce[c * 64 + d], Xc[t]);
  if (d == 0) atomicAdd(&cnt[c], 1.f);
}
__global__ void pool_div_k(float* ce, const float* cnt){
  int t = blockIdx.x * blockDim.x + threadIdx.x;
  if (t < 64 * 64) ce[t] /= fmaxf(cnt[t >> 6], 1.f);
}

// ---------------- GAT pieces (shared by cluster-mode and node-mode) ----------------
__global__ void att_k(const float* xh, const float* atts, const float* attd,
                      float* as_o, float* ad_o, int n){
  int t = blockIdx.x * blockDim.x + threadIdx.x;
  if (t >= 2 * n) return;
  int i = t >> 1, h = t & 1;
  float s = 0.f, d2 = 0.f;
  for (int d = 0; d < 64; ++d){
    float x = xh[(long long)i * 128 + h * 64 + d];
    s  += x * atts[h * 64 + d];
    d2 += x * attd[h * 64 + d];
  }
  as_o[i * 2 + h] = s;
  ad_o[i * 2 + h] = d2;
}
// Me[k*2+h] = sum_d we[k,h*64+d]*atte[h,d]  (collapses ea@We contraction to 2x2)
__global__ void me_k(const float* we, const float* atte, float* me){
  int t = threadIdx.x;
  if (t >= 4) return;
  int k = t >> 1, h = t & 1;
  float s = 0.f;
  for (int d = 0; d < 64; ++d) s += we[k * 128 + h * 64 + d] * atte[h * 64 + d];
  me[k * 2 + h] = s;
}
__global__ void stats_k(const int* src, const int* dst, const int* assign, const float* ea,
                        float* cnt, float* easum, int E, int cmode){
  int e = blockIdx.x * blockDim.x + threadIdx.x;
  if (e >= E) return;
  int d = dst[e];
  if (cmode){ int cs = assign[src[e]], cd = assign[d]; if (cs == cd) return; d = cd; }
  atomicAdd(&cnt[d], 1.f);
  atomicAdd(&easum[d * 2 + 0], ea[(long long)e * 2 + 0]);
  atomicAdd(&easum[d * 2 + 1], ea[(long long)e * 2 + 1]);
}
__global__ void loop_k(const float* as, const float* ad, const float* easum, const float* cnt,
                       const float* me, float* lgl, float* mx, int n){
  int t = blockIdx.x * blockDim.x + threadIdx.x;
  if (t >= 2 * n) return;
  int i = t >> 1, h = t & 1;
  float c = fmaxf(cnt[i], 1.f);
  float m0 = easum[i * 2 + 0] / c, m1 = easum[i * 2 + 1] / c;
  float lg = leaky02(as[t] + ad[t] + m0 * me[h] + m1 * me[2 + h]);
  lgl[t] = lg;
  mx[t]  = lg;
}
__global__ void edge_max_k(const int* src, const int* dst, const int* assign, const float* ea,
                           const float* as, const float* ad, const float* me, float* mx,
                           int E, int cmode){
  int t = blockIdx.x * blockDim.x + threadIdx.x;
  if (t >= 2 * E) return;
  int e = t >> 1, h = t & 1;
  int s = src[e], d = dst[e];
  if (cmode){ int cs = assign[s], cd = assign[d]; if (cs == cd) return; s = cs; d = cd; }
  float lg = leaky02(as[s * 2 + h] + ad[d * 2 + h] +
                     ea[(long long)e * 2 + 0] * me[h] + ea[(long long)e * 2 + 1] * me[2 + h]);
  atomicMaxF(&mx[d * 2 + h], lg);
}
__global__ void exl_k(const float* lgl, const float* mx, float* exl, float* den, int n){
  int t = blockIdx.x * blockDim.x + threadIdx.x;
  if (t >= 2 * n) return;
  float v = expf(lgl[t] - mx[t]);
  exl[t] = v;
  den[t] = v;
}
__global__ void edge_den_k(const int* src, const int* dst, const int* assign, const float* ea,
                           const float* as, const float* ad, const float* me, const float* mx,
                           float* den, float* exbuf, int E, int cmode){
  int t = blockIdx.x * blockDim.x + threadIdx.x;
  if (t >= 2 * E) return;
  int e = t >> 1, h = t & 1;
  int s = src[e], d = dst[e];
  if (cmode){
    int cs = assign[s], cd = assign[d];
    if (cs == cd){ exbuf[(long long)e * 2 + h] = 0.f; return; }
    s = cs; d = cd;
  }
  float lg = leaky02(as[s * 2 + h] + ad[d * 2 + h] +
                     ea[(long long)e * 2 + 0] * me[h] + ea[(long long)e * 2 + 1] * me[2 + h]);
  float ex = expf(lg - mx[d * 2 + h]);
  exbuf[(long long)e * 2 + h] = ex;
  atomicAdd(&den[d * 2 + h], ex);
}
__global__ void edge_out_k(const int* src, const int* dst, const int* assign,
                           const float* exbuf, const float* den, const float* xh,
                           float* out, int E, int cmode){
  int e = blockIdx.x * 2 + (threadIdx.x >> 7);
  int f = threadIdx.x & 127;     // h*64 + d
  if (e >= E) return;
  int h = f >> 6;
  float ex = exbuf[(long long)e * 2 + h];
  if (ex == 0.f) return;
  int s = src[e], d = dst[e];
  if (cmode){ s = assign[s]; d = assign[d]; }
  float coef = ex / den[d * 2 + h];
  atomicAdd(&out[(long long)d * 128 + f], coef * xh[(long long)s * 128 + f]);
}
__global__ void final_head_k(const float* out, const float* exl, const float* den,
                             const float* xh, const float* bias, float* res, int n){
  long long t = (long long)blockIdx.x * blockDim.x + threadIdx.x;
  if (t >= (long long)n * 64) return;
  int i = (int)(t >> 6), d = (int)(t & 63);
  float r = 0.f;
#pragma unroll
  for (int h = 0; h < 2; ++h){
    float sl = exl[i * 2 + h] / den[i * 2 + h];
    r += out[(long long)i * 128 + h * 64 + d] + sl * xh[(long long)i * 128 + h * 64 + d];
  }
  res[t] = 0.5f * r + bias[d];
}

// ---------------- combine / norm ----------------
__global__ void combine_k(const float* Xc, const float* cu, const int* assign, float* X, int n){
  long long t = (long long)blockIdx.x * blockDim.x + threadIdx.x;
  if (t >= (long long)n * 64) return;
  int node = (int)(t >> 6), d = (int)(t & 63);
  X[t] = Xc[t] + cu[assign[node] * 64 + d];
}
__global__ void colsum_k(const float* X, float* out, int nrows){
  __shared__ float sh[256];
  int d = blockIdx.x;
  int stride = gridDim.y * blockDim.x;
  float s = 0.f;
  for (int r = blockIdx.y * blockDim.x + threadIdx.x; r < nrows; r += stride){
    s += X[(long long)r * 64 + d];
    __builtin_prefetch(&X[(long long)(r + stride) * 64 + d], 0, 0); // global_prefetch_b8
  }
  sh[threadIdx.x] = s; __syncthreads();
  for (int off = 128; off > 0; off >>= 1){
    if ((int)threadIdx.x < off) sh[threadIdx.x] += sh[threadIdx.x + off];
    __syncthreads();
  }
  if (threadIdx.x == 0) atomicAdd(&out[d], sh[0]);
}
__global__ void colsum_sq_k(const float* X, const float* musum, const float* nms,
                            float* out, int nrows){
  __shared__ float sh[256];
  int d = blockIdx.x;
  float mu = musum[d] / (float)nrows;
  float nm = nms[d];
  int stride = gridDim.y * blockDim.x;
  float s = 0.f;
  for (int r = blockIdx.y * blockDim.x + threadIdx.x; r < nrows; r += stride){
    float c = X[(long long)r * 64 + d] - nm * mu;
    s += c * c;
    __builtin_prefetch(&X[(long long)(r + stride) * 64 + d], 0, 0);
  }
  sh[threadIdx.x] = s; __syncthreads();
  for (int off = 128; off > 0; off >>= 1){
    if ((int)threadIdx.x < off) sh[threadIdx.x] += sh[threadIdx.x + off];
    __syncthreads();
  }
  if (threadIdx.x == 0) atomicAdd(&out[d], sh[0]);
}
__global__ void norm_elu_k(const float* Xp, const float* Xin, const float* musum,
                           const float* varsum, const float* nw, const float* nb,
                           const float* nms, float* Xout, int nrows){
  long long t = (long long)blockIdx.x * blockDim.x + threadIdx.x;
  if (t >= (long long)nrows * 64) return;
  int d = (int)(t & 63);
  float invN = 1.f / (float)nrows;
  float mu = musum[d] * invN;
  float ctr = Xp[t] - nms[d] * mu;
  float y = nw[d] * ctr * rsqrtf(varsum[d] * invN + 1e-5f) + nb[d] + Xin[t];
  Xout[t] = y > 0.f ? y : (expf(y) - 1.f);
}

// ---------------- host ----------------
extern "C" void kernel_launch(void* const* d_in, const int* in_sizes, int n_in,
                              void* d_out, int out_size, void* d_ws, size_t ws_size,
                              hipStream_t stream){
  (void)in_sizes; (void)n_in; (void)out_size; (void)ws_size;
  const float* extra    = (const float*)d_in[0];
  const float* ea       = (const float*)d_in[1];
  const float* user_emb = (const float*)d_in[2];
  const float* item_emb = (const float*)d_in[3];
  const float* fuse_w1  = (const float*)d_in[4];
  const float* fuse_b1  = (const float*)d_in[5];
  const float* fuse_w2  = (const float*)d_in[6];
  const float* fuse_b2  = (const float*)d_in[7];
  const float* gcn_w    = (const float*)d_in[8];
  const float* gcn_b    = (const float*)d_in[9];
  const float* gw       = (const float*)d_in[10];
  const float* gatt_s   = (const float*)d_in[11];
  const float* gatt_d   = (const float*)d_in[12];
  const float* gwe      = (const float*)d_in[13];
  const float* gatt_e   = (const float*)d_in[14];
  const float* gb       = (const float*)d_in[15];
  const float* lw       = (const float*)d_in[16];
  const float* latt_s   = (const float*)d_in[17];
  const float* latt_d   = (const float*)d_in[18];
  const float* lwe      = (const float*)d_in[19];
  const float* latt_e   = (const float*)d_in[20];
  const float* lb       = (const float*)d_in[21];
  const float* nw       = (const float*)d_in[22];
  const float* nb       = (const float*)d_in[23];
  const float* nms      = (const float*)d_in[24];
  const int*   assign   = (const int*)d_in[25];
  const int*   ei       = (const int*)d_in[26];
  const int* src = ei;
  const int* dst = ei + EE;
  float* out_f = (float*)d_out;

  // workspace bump allocator
  float* ws = (float*)d_ws;
  size_t o = 0;
  auto alloc = [&](size_t n){ float* p = ws + o; o += n; return p; };
  float* B_X   = alloc((size_t)NN * 64);   // X / X_comb
  float* B_h   = alloc((size_t)NN * 64);   // GCN h / Xin residual
  float* B_Xc  = alloc((size_t)NN * 64);   // GCN agg->Xc / prenorm
  float* B_T1  = alloc((size_t)NN * 128);  // fuse hidden / xh
  float* B_T2  = alloc((size_t)NN * 128);  // out accumulator [N,2,64]
  float* B_deg = alloc((size_t)NN);        // deg/dinv/cnt
  float* B_as  = alloc((size_t)NN * 2);
  float* B_ad  = alloc((size_t)NN * 2);
  float* B_mx  = alloc((size_t)NN * 2);
  float* B_den = alloc((size_t)NN * 2);
  float* B_exl = alloc((size_t)NN * 2);
  float* B_lgl = alloc((size_t)NN * 2);
  float* B_eam = alloc((size_t)NN * 2);
  float* B_ex  = alloc((size_t)EE * 2);
  float* B_ce  = alloc(64 * 64);
  float* B_xhc = alloc(64 * 128);
  float* B_cntc= alloc(64);
  float* B_asc = alloc(128);
  float* B_adc = alloc(128);
  float* B_cntg= alloc(64);
  float* B_eag = alloc(128);
  float* B_mxg = alloc(128);
  float* B_deng= alloc(128);
  float* B_exlg= alloc(128);
  float* B_lglg= alloc(128);
  float* B_outc= alloc(64 * 128);
  float* B_cu  = alloc(64 * 64);
  float* B_me  = alloc(4);
  float* B_mu  = alloc(64);
  float* B_var = alloc(64);
  // f16 transposed weights (49152 halves = 24576 floats)
  _Float16* Hbase = (_Float16*)alloc(24576);
  _Float16* H_fw1 = Hbase;           // 96*128
  _Float16* H_fw2 = Hbase + 12288;   // 128*64
  _Float16* H_gcn = Hbase + 20480;   // 64*64
  _Float16* H_gw  = Hbase + 24576;   // 64*128
  _Float16* H_lw0 = Hbase + 32768;   // 64*128
  _Float16* H_lw1 = Hbase + 40960;   // 64*128

  auto cdiv = [](long long a, long long b){ return (int)((a + b - 1) / b); };

  // ---- weight convert/transpose to f16 (tiny) ----
  convT_k<<<cdiv(96 * 128, 256), 256, 0, stream>>>(fuse_w1, H_fw1, 96, 128);
  convT_k<<<cdiv(128 * 64, 256), 256, 0, stream>>>(fuse_w2, H_fw2, 128, 64);
  convT_k<<<cdiv(64 * 64, 256), 256, 0, stream>>>(gcn_w, H_gcn, 64, 64);
  convT_k<<<cdiv(64 * 128, 256), 256, 0, stream>>>(gw, H_gw, 64, 128);
  convT_k<<<cdiv(64 * 128, 256), 256, 0, stream>>>(lw, H_lw0, 64, 128);
  convT_k<<<cdiv(64 * 128, 256), 256, 0, stream>>>(lw + 64 * 128, H_lw1, 64, 128);

  // ---- Stage A: fused item features, assemble X ----
  wmma_gemm<1><<<dim3(NU / 16, 2), 32, 0, stream>>>(item_emb, 96, extra, H_fw1, fuse_b1, B_T1, 128, 96, 1);
  wmma_gemm<0><<<dim3(NU / 16, 1), 32, 0, stream>>>(B_T1, 128, nullptr, H_fw2, fuse_b2, B_X + (size_t)NU * 64, 64, 128, 0);
  copy_k<<<cdiv((long long)NU * 64, 256), 256, 0, stream>>>(B_X, user_emb, (long long)NU * 64);

  // ---- Stage B: GCN over intra-cluster edges ----
  wmma_gemm<0><<<dim3(NN / 16, 1), 32, 0, stream>>>(B_X, 64, nullptr, H_gcn, nullptr, B_h, 64, 64, 0);
  fill_k<<<cdiv(NN, 256), 256, 0, stream>>>(B_deg, 1.f, NN);
  gcn_deg_k<<<cdiv(EE, 256), 256, 0, stream>>>(src, dst, assign, B_deg, EE);
  rsqrt_k<<<cdiv(NN, 256), 256, 0, stream>>>(B_deg, NN);
  fill_k<<<cdiv((long long)NN * 64, 256), 256, 0, stream>>>(B_Xc, 0.f, (long long)NN * 64);
  gcn_agg_k<<<cdiv(EE, 4), 256, 0, stream>>>(src, dst, assign, B_deg, B_h, B_Xc, EE);
  gcn_final_k<<<cdiv((long long)NN * 64, 256), 256, 0, stream>>>(B_Xc, B_deg, B_h, gcn_b, NN);

  // ---- Stage C: cluster mean pooling ----
  fill_k<<<cdiv(64 * 64, 256), 256, 0, stream>>>(B_ce, 0.f, 64 * 64);
  fill_k<<<1, 64, 0, stream>>>(B_cntc, 0.f, 64);
  pool_k<<<cdiv((long long)NN * 64, 256), 256, 0, stream>>>(B_Xc, assign, B_ce, B_cntc, NN);
  pool_div_k<<<16, 256, 0, stream>>>(B_ce, B_cntc);

  // ---- Stage D: cluster-level GAT over inter-cluster edges ----
  wmma_gemm<0><<<dim3(4, 2), 32, 0, stream>>>(B_ce, 64, nullptr, H_gw, nullptr, B_xhc, 128, 64, 0);
  att_k<<<1, 128, 0, stream>>>(B_xhc, gatt_s, gatt_d, B_asc, B_adc, NC);
  me_k<<<1, 4, 0, stream>>>(gwe, gatt_e, B_me);
  fill_k<<<1, 64, 0, stream>>>(B_cntg, 0.f, 64);
  fill_k<<<1, 128, 0, stream>>>(B_eag, 0.f, 128);
  stats_k<<<cdiv(EE, 256), 256, 0, stream>>>(src, dst, assign, ea, B_cntg, B_eag, EE, 1);
  loop_k<<<1, 128, 0, stream>>>(B_asc, B_adc, B_eag, B_cntg, B_me, B_lglg, B_mxg, NC);
  edge_max_k<<<cdiv(2LL * EE, 256), 256, 0, stream>>>(src, dst, assign, ea, B_asc, B_adc, B_me, B_mxg, EE, 1);
  exl_k<<<1, 128, 0, stream>>>(B_lglg, B_mxg, B_exlg, B_deng, NC);
  edge_den_k<<<cdiv(2LL * EE, 256), 256, 0, stream>>>(src, dst, assign, ea, B_asc, B_adc, B_me, B_mxg, B_deng, B_ex, EE, 1);
  fill_k<<<cdiv(64 * 128, 256), 256, 0, stream>>>(B_outc, 0.f, 64 * 128);
  edge_out_k<<<cdiv(EE, 2), 256, 0, stream>>>(src, dst, assign, B_ex, B_deng, B_xhc, B_outc, EE, 1);
  final_head_k<<<16, 256, 0, stream>>>(B_outc, B_exlg, B_deng, B_xhc, gb, B_cu, NC);

  // ---- Stage E: X_comb = Xc + cu[assign] ----
  combine_k<<<cdiv((long long)NN * 64, 256), 256, 0, stream>>>(B_Xc, B_cu, assign, B_X, NN);

  // ---- Stage F: node-level GAT layers + GraphNorm + ELU residual ----
  for (int l = 0; l < 2; ++l){
    copy_k<<<cdiv((long long)NN * 64, 256), 256, 0, stream>>>(B_h, B_X, (long long)NN * 64); // Xin
    wmma_gemm<0><<<dim3(NN / 16, 2), 32, 0, stream>>>(B_X, 64, nullptr, (l ? H_lw1 : H_lw0), nullptr, B_T1, 128, 64, 0);
    att_k<<<cdiv(2LL * NN, 256), 256, 0, stream>>>(B_T1, latt_s + l * 128, latt_d + l * 128, B_as, B_ad, NN);
    me_k<<<1, 4, 0, stream>>>(lwe + l * 256, latt_e + l * 128, B_me);
    fill_k<<<cdiv(NN, 256), 256, 0, stream>>>(B_deg, 0.f, NN);
    fill_k<<<cdiv(2LL * NN, 256), 256, 0, stream>>>(B_eam, 0.f, 2LL * NN);
    stats_k<<<cdiv(EE, 256), 256, 0, stream>>>(src, dst, assign, ea, B_deg, B_eam, EE, 0);
    loop_k<<<cdiv(2LL * NN, 256), 256, 0, stream>>>(B_as, B_ad, B_eam, B_deg, B_me, B_lgl, B_mx, NN);
    edge_max_k<<<cdiv(2LL * EE, 256), 256, 0, stream>>>(src, dst, assign, ea, B_as, B_ad, B_me, B_mx, EE, 0);
    exl_k<<<cdiv(2LL * NN, 256), 256, 0, stream>>>(B_lgl, B_mx, B_exl, B_den, NN);
    edge_den_k<<<cdiv(2LL * EE, 256), 256, 0, stream>>>(src, dst, assign, ea, B_as, B_ad, B_me, B_mx, B_den, B_ex, EE, 0);
    fill_k<<<cdiv((long long)NN * 128, 256), 256, 0, stream>>>(B_T2, 0.f, (long long)NN * 128);
    edge_out_k<<<cdiv(EE, 2), 256, 0, stream>>>(src, dst, assign, B_ex, B_den, B_T1, B_T2, EE, 0);
    final_head_k<<<cdiv((long long)NN * 64, 256), 256, 0, stream>>>(B_T2, B_exl, B_den, B_T1, lb + l * 64, B_Xc, NN);
    // GraphNorm + residual + ELU
    fill_k<<<1, 64, 0, stream>>>(B_mu, 0.f, 64);
    colsum_k<<<dim3(64, 32), 256, 0, stream>>>(B_Xc, B_mu, NN);
    fill_k<<<1, 64, 0, stream>>>(B_var, 0.f, 64);
    colsum_sq_k<<<dim3(64, 32), 256, 0, stream>>>(B_Xc, B_mu, nms + l * 64, B_var, NN);
    norm_elu_k<<<cdiv((long long)NN * 64, 256), 256, 0, stream>>>(B_Xc, B_h, B_mu, B_var, nw + l * 64, nb + l * 64, nms + l * 64, B_X, NN);
  }

  // output = X_comb (users rows 0..NU-1, items rows NU..NN-1, already in order)
  copy_k<<<cdiv((long long)NN * 64, 256), 256, 0, stream>>>(out_f, B_X, (long long)NN * 64);
}
